// VectorQuantizerST_14912126451769
// MI455X (gfx1250) — matrix operand order repaired and checked by reading, not compiled, not run on hardware
//
#include <hip/hip_runtime.h>
#include <hip/hip_bf16.h>

typedef __attribute__((ext_vector_type(16))) __bf16          v16bf;
typedef __attribute__((ext_vector_type(8)))  float           v8f;
typedef __attribute__((ext_vector_type(8)))  unsigned short  v8us;
typedef __attribute__((ext_vector_type(16))) unsigned short  v16us;

#define N_ROWS 32768
#define KCODES 8192
#define DIM    256
#define STAGE  32                      // codes per LDS stage (2 WMMA column tiles)

__device__ __forceinline__ unsigned short f32_to_bf16_rne(float f) {
    unsigned int u = __float_as_uint(f);
    unsigned int r = u + 0x7FFFu + ((u >> 16) & 1u);   // round-to-nearest-even
    return (unsigned short)(r >> 16);
}

// ---------------------------------------------------------------------------
// Kernel 0: zero histogram + loss accumulator (must re-run every graph replay)
// ---------------------------------------------------------------------------
__global__ void vq_zero_kernel(unsigned int* __restrict__ counts,
                               float* __restrict__ loss_acc) {
    int i = blockIdx.x * 256 + threadIdx.x;
    if (i < KCODES) counts[i] = 0u;
    if (i == 0)     loss_acc[0] = 0.0f;
}

// ---------------------------------------------------------------------------
// Kernel 1: codebook f32 -> bf16 + ||e||^2.  One wave per code row.
// ---------------------------------------------------------------------------
__global__ __launch_bounds__(256) void vq_prep_kernel(
    const float* __restrict__ cb_f32,
    unsigned short* __restrict__ cb_bf16,
    float* __restrict__ e_sq) {
    const int lane = threadIdx.x & 31;
    const int wave = threadIdx.x >> 5;
    const int row  = blockIdx.x * 8 + wave;          // 1024 blocks * 8 waves = 8192
    const float* src = cb_f32 + (size_t)row * DIM + lane * 8;
    const float4 f0 = *(const float4*)(src);
    const float4 f1 = *(const float4*)(src + 4);
    v8us o;
    o[0] = f32_to_bf16_rne(f0.x); o[1] = f32_to_bf16_rne(f0.y);
    o[2] = f32_to_bf16_rne(f0.z); o[3] = f32_to_bf16_rne(f0.w);
    o[4] = f32_to_bf16_rne(f1.x); o[5] = f32_to_bf16_rne(f1.y);
    o[6] = f32_to_bf16_rne(f1.z); o[7] = f32_to_bf16_rne(f1.w);
    *(v8us*)(cb_bf16 + (size_t)row * DIM + lane * 8) = o;
    float s = f0.x*f0.x + f0.y*f0.y + f0.z*f0.z + f0.w*f0.w
            + f1.x*f1.x + f1.y*f1.y + f1.z*f1.z + f1.w*f1.w;
    #pragma unroll
    for (int off = 16; off >= 1; off >>= 1) s += __shfl_down(s, off, 32);
    if (lane == 0) e_sq[row] = s;
}

// ---------------------------------------------------------------------------
// Kernel 2: fused GEMM + argmin via v_wmma_f32_16x16x32_bf16.
// Block = 8 waves = 128 z-rows.  Codebook streamed through a double-buffered
// LDS stage (32 codes = 16 KiB bf16) using GLOBAL_LOAD_ASYNC_TO_LDS_B128
// (no VGPR round trip, tracked by ASYNCcnt).  Two independent accumulators
// per wave break the WMMA C-dependency chain.
// ---------------------------------------------------------------------------
__global__ __launch_bounds__(256) void vq_argmin_kernel(
    const float* __restrict__ z,
    const unsigned short* __restrict__ cb,   // bf16 codebook
    const float* __restrict__ e_sq,
    float* __restrict__ idx_out,             // N floats (indices as float)
    unsigned int* __restrict__ counts) {
    __shared__ unsigned short lds_cb[2][STAGE * DIM];   // 2 x 16 KiB

    const int tid  = threadIdx.x;
    const int lane = tid & 31;
    const int wave = tid >> 5;
    const int m    = lane & 15;              // column / A-row within tile
    const int h    = lane >> 4;              // half-wave
    const int rowBase = blockIdx.x * 128 + wave * 16;

    // Build the 8 A fragments (16x32 bf16 each) once; held in VGPRs.
    // Layout: lane m,h holds K = 32c + 16h + [0,16) (matches B fragments).
    const float* zrow = z + (size_t)(rowBase + m) * DIM;
    v16us afrag[8];
    #pragma unroll
    for (int c = 0; c < 8; ++c) {
        const int k0 = c * 32 + 16 * h;
        const float4 f0 = *(const float4*)(zrow + k0);
        const float4 f1 = *(const float4*)(zrow + k0 + 4);
        const float4 f2 = *(const float4*)(zrow + k0 + 8);
        const float4 f3 = *(const float4*)(zrow + k0 + 12);
        v16us a;
        a[0]  = f32_to_bf16_rne(f0.x); a[1]  = f32_to_bf16_rne(f0.y);
        a[2]  = f32_to_bf16_rne(f0.z); a[3]  = f32_to_bf16_rne(f0.w);
        a[4]  = f32_to_bf16_rne(f1.x); a[5]  = f32_to_bf16_rne(f1.y);
        a[6]  = f32_to_bf16_rne(f1.z); a[7]  = f32_to_bf16_rne(f1.w);
        a[8]  = f32_to_bf16_rne(f2.x); a[9]  = f32_to_bf16_rne(f2.y);
        a[10] = f32_to_bf16_rne(f2.z); a[11] = f32_to_bf16_rne(f2.w);
        a[12] = f32_to_bf16_rne(f3.x); a[13] = f32_to_bf16_rne(f3.y);
        a[14] = f32_to_bf16_rne(f3.z); a[15] = f32_to_bf16_rne(f3.w);
        afrag[c] = a;
    }

    float best[8];
    int   bidx[8];
    #pragma unroll
    for (int j = 0; j < 8; ++j) { best[j] = 3.4e38f; bidx[j] = 0; }

    // Async stage copy: each thread DMAs its 64 B slice (4 x b128) of stage s
    // straight into LDS.  Per-lane LDS byte address = low 32 bits of the
    // generic pointer (flat aperture).
    auto issue_stage = [&](int s) {
        const unsigned short* gp =
            cb + (size_t)s * (STAGE * DIM) + (size_t)tid * 32;
        unsigned long long g = (unsigned long long)gp;
        unsigned l = (unsigned)(size_t)(&lds_cb[s & 1][0] + 32 * tid);
        asm volatile("global_load_async_to_lds_b128 %0, %1, off"
                     :: "v"(l), "v"(g) : "memory");
        asm volatile("global_load_async_to_lds_b128 %0, %1, off offset:16"
                     :: "v"(l), "v"(g) : "memory");
        asm volatile("global_load_async_to_lds_b128 %0, %1, off offset:32"
                     :: "v"(l), "v"(g) : "memory");
        asm volatile("global_load_async_to_lds_b128 %0, %1, off offset:48"
                     :: "v"(l), "v"(g) : "memory");
    };

    issue_stage(0);

    const int NSTAGE = KCODES / STAGE;       // 256
    for (int s = 0; s < NSTAGE; ++s) {
        // Wait own async DMAs (syncthreads does not cover ASYNCcnt), then
        // barrier so every wave's slice of stage s is visible.
        asm volatile("s_wait_asynccnt 0x0" ::: "memory");
        __syncthreads();
        // Prefetch stage s+1 into the other buffer: all waves have passed the
        // barrier above, so all stage s-1 reads of that buffer are retired.
        if (s + 1 < NSTAGE) issue_stage(s + 1);

        const unsigned short* buf = lds_cb[s & 1];
        // B fragments (32x16 bf16): lane = column m, 16 contiguous K at 16*h.
        const unsigned short* b0p = buf + m * DIM + 16 * h;
        const unsigned short* b1p = b0p + 16 * DIM;
        v8f acc0 = {0.f,0.f,0.f,0.f,0.f,0.f,0.f,0.f};
        v8f acc1 = {0.f,0.f,0.f,0.f,0.f,0.f,0.f,0.f};
        #pragma unroll
        for (int c = 0; c < 8; ++c) {
            v16us b0 = *(const v16us*)(b0p + c * 32);
            v16us b1 = *(const v16us*)(b1p + c * 32);
            acc0 = __builtin_amdgcn_wmma_f32_16x16x32_bf16(
                false, __builtin_bit_cast(v16bf, afrag[c]),
                false, __builtin_bit_cast(v16bf, b0),
                (short)0, acc0, false, false);
            acc1 = __builtin_amdgcn_wmma_f32_16x16x32_bf16(
                false, __builtin_bit_cast(v16bf, afrag[c]),
                false, __builtin_bit_cast(v16bf, b1),
                (short)0, acc1, false, false);
        }
        const int n0 = s * STAGE + m;        // code scored by acc0 on this lane
        const int n1 = n0 + 16;              // code scored by acc1
        const float es0 = e_sq[n0];
        const float es1 = e_sq[n1];
        #pragma unroll
        for (int j = 0; j < 8; ++j) {
            // ||z||^2 constant per row: argmin over es - 2*dot
            float s0 = __builtin_fmaf(-2.0f, acc0[j], es0);
            if (s0 < best[j]) { best[j] = s0; bidx[j] = n0; }
            float s1 = __builtin_fmaf(-2.0f, acc1[j], es1);
            if (s1 < best[j]) { best[j] = s1; bidx[j] = n1; }
        }
        __syncthreads();   // all waves done reading buf before next overwrite
    }

    // Reduce across the 16 lanes that share each output row (xor <= 8 stays
    // inside a 16-lane half). Tie-break to the lowest index (argmin semantics).
    #pragma unroll
    for (int off = 1; off <= 8; off <<= 1) {
        #pragma unroll
        for (int j = 0; j < 8; ++j) {
            float os = __shfl_xor(best[j], off, 32);
            int   oi = __shfl_xor(bidx[j], off, 32);
            if (os < best[j] || (os == best[j] && oi < bidx[j])) {
                best[j] = os; bidx[j] = oi;
            }
        }
    }
    if (m == 0) {
        // C/D layout: VGPR j, half h -> row M = 8*h + j
        #pragma unroll
        for (int j = 0; j < 8; ++j) {
            const int row = rowBase + 8 * h + j;
            idx_out[row] = (float)bidx[j];
            atomicAdd(&counts[bidx[j]], 1u);
        }
    }
}

// ---------------------------------------------------------------------------
// Kernel 3: gather z_q from the f32 codebook (exact), write z_q_st, SSE.
// ---------------------------------------------------------------------------
__global__ __launch_bounds__(256) void vq_gather_loss_kernel(
    const float* __restrict__ z,
    const float* __restrict__ cb_f32,
    const float* __restrict__ idx_out,
    float* __restrict__ zq_out,
    float* __restrict__ loss_acc) {
    __shared__ float red[8];
    const int row = blockIdx.x;
    const int d   = threadIdx.x;
    const int idx = (int)idx_out[row];
    const float q  = cb_f32[(size_t)idx * DIM + d];
    const float ze = z[(size_t)row * DIM + d];
    zq_out[(size_t)row * DIM + d] = q;     // z_q_st == z_q numerically
    float diff = ze - q;
    float sq = diff * diff;
    #pragma unroll
    for (int off = 16; off >= 1; off >>= 1) sq += __shfl_down(sq, off, 32);
    if ((d & 31) == 0) red[d >> 5] = sq;
    __syncthreads();
    if (d == 0) {
        float s = 0.0f;
        #pragma unroll
        for (int w = 0; w < 8; ++w) s += red[w];
        atomicAdd(loss_acc, s);
    }
}

// ---------------------------------------------------------------------------
// Kernel 4: loss_vq = 1.25 * mean, perplexity from histogram.
// ---------------------------------------------------------------------------
__global__ __launch_bounds__(256) void vq_finalize_kernel(
    const unsigned int* __restrict__ counts,
    const float* __restrict__ loss_acc,
    float* __restrict__ loss_out,
    float* __restrict__ perp_out) {
    __shared__ float red[256];
    float ent = 0.0f;
    for (int b = threadIdx.x; b < KCODES; b += 256) {
        float p = (float)counts[b] * (1.0f / (float)N_ROWS);
        ent += p * logf(p + 1e-12f);
    }
    red[threadIdx.x] = ent;
    __syncthreads();
    for (int s = 128; s > 0; s >>= 1) {
        if (threadIdx.x < s) red[threadIdx.x] += red[threadIdx.x + s];
        __syncthreads();
    }
    if (threadIdx.x == 0) {
        loss_out[0] = loss_acc[0] * (1.25f / (float)(N_ROWS * DIM));
        perp_out[0] = expf(-red[0]);
    }
}

// ---------------------------------------------------------------------------
extern "C" void kernel_launch(void* const* d_in, const int* in_sizes, int n_in,
                              void* d_out, int out_size, void* d_ws, size_t ws_size,
                              hipStream_t stream) {
    const float* z  = (const float*)d_in[0];      // [32,32,32,256] f32
    const float* cb = (const float*)d_in[1];      // [8192,256] f32

    float* out      = (float*)d_out;
    float* zq_out   = out;                        // 8388608 floats
    float* idx_out  = out + (size_t)N_ROWS * DIM; // 32768 floats
    float* loss_out = idx_out + N_ROWS;           // 1 float
    float* perp_out = loss_out + 1;               // 1 float

    unsigned char* ws = (unsigned char*)d_ws;
    unsigned short* cb_bf16 = (unsigned short*)ws;                 // 4 MiB
    float*        e_sq      = (float*)(ws + 4194304);              // 32 KiB
    unsigned int* counts    = (unsigned int*)(ws + 4227072);       // 32 KiB
    float*        loss_acc  = (float*)(ws + 4259840);              // 4 B

    vq_zero_kernel<<<(KCODES + 255) / 256, 256, 0, stream>>>(counts, loss_acc);
    vq_prep_kernel<<<KCODES / 8, 256, 0, stream>>>(cb, cb_bf16, e_sq);
    vq_argmin_kernel<<<N_ROWS / 128, 256, 0, stream>>>(z, cb_bf16, e_sq,
                                                       idx_out, counts);
    vq_gather_loss_kernel<<<N_ROWS, 256, 0, stream>>>(z, cb, idx_out,
                                                      zq_out, loss_acc);
    vq_finalize_kernel<<<1, 256, 0, stream>>>(counts, loss_acc,
                                              loss_out, perp_out);
}
